// SupConLossWithPrototype_816043786681
// MI455X (gfx1250) — matrix operand order repaired and checked by reading, not compiled, not run on hardware
//
#include <hip/hip_runtime.h>

// ---------------------------------------------------------------------------
// SupCon-with-prototypes loss, fused for MI455X (gfx1250, wave32, WMMA).
//
// Pipeline (all deterministic, all on `stream`):
//   k_flags      : masks + deterministic scan-compaction of novel indices
//   k_convert    : f32 -> (f16 hi, f16 residual) split of features
//   k_gather     : gather novel rows into contiguous Fn (zero-padded to 16)
//   k_gram       : F @ Fn^T via v_wmma_f32_16x16x32_f16 (3-way split product),
//                  fused per-row sum(exp(sim)) and sum(sim) reductions
//   k_rowcombine : prototype dots, diagonal correction, per-row loss terms
//   k_final      : fixed-order tree reduction + gating -> scalar
// ---------------------------------------------------------------------------

typedef _Float16 v16h __attribute__((ext_vector_type(16)));
typedef _Float16 v8h  __attribute__((ext_vector_type(8)));
typedef float    v8f  __attribute__((ext_vector_type(8)));

#define M_ROWS      8192
#define K_DIM       128
#define B_PROTOS    60
#define NUM_CLASSES 80
#define INV_T       5.0f   // 1 / TEMPERATURE

// ---- workspace layout (256B aligned regions) ------------------------------
constexpr size_t alignUp(size_t x) { return (x + 255) & ~size_t(255); }
constexpr size_t OFF_CNT = 0;                                              // int[16]
constexpr size_t OFF_IDX = alignUp(OFF_CNT + 16 * sizeof(int));            // int[M]
constexpr size_t OFF_NF  = alignUp(OFF_IDX + M_ROWS * sizeof(int));        // char[M]
constexpr size_t OFF_BF  = alignUp(OFF_NF + M_ROWS);                       // char[M]
constexpr size_t OFF_E   = alignUp(OFF_BF + M_ROWS);                       // float[M]
constexpr size_t OFF_S   = alignUp(OFF_E + M_ROWS * sizeof(float));        // float[M]
constexpr size_t OFF_CN  = alignUp(OFF_S + M_ROWS * sizeof(float));        // float[M]
constexpr size_t OFF_CB  = alignUp(OFF_CN + M_ROWS * sizeof(float));       // float[M]
constexpr size_t OFF_FH  = alignUp(OFF_CB + M_ROWS * sizeof(float));       // f16[M*K]
constexpr size_t OFF_FL  = alignUp(OFF_FH + M_ROWS * K_DIM * 2);           // f16[M*K]
constexpr size_t OFF_GH  = alignUp(OFF_FL + M_ROWS * K_DIM * 2);           // f16[M*K]
constexpr size_t OFF_GL  = alignUp(OFF_GH + M_ROWS * K_DIM * 2);           // f16[M*K]

// ---------------------------------------------------------------------------
// 1) flags + deterministic compaction of novel indices (single workgroup)
// ---------------------------------------------------------------------------
__global__ __launch_bounds__(1024)
void k_flags(const int* __restrict__ labels, int* __restrict__ counters,
             int* __restrict__ novel_idx, char* __restrict__ nf,
             char* __restrict__ bf) {
  __shared__ int sn[1024];
  __shared__ int sf[1024];
  __shared__ int sb[1024];
  const int t = threadIdx.x;
  int lnov = 0, lfg = 0, lbase = 0;
  int flag[8];
#pragma unroll
  for (int r = 0; r < 8; ++r) {
    const int row = t * 8 + r;
    const int lab = labels[row];
    const bool fg    = (lab != NUM_CLASSES);
    const bool base  = fg && (lab >= 0) && (lab < B_PROTOS);  // proto_labels == arange(B)
    const bool novel = fg && !base;
    nf[row] = novel ? 1 : 0;
    bf[row] = base ? 1 : 0;
    flag[r] = novel ? 1 : 0;
    lnov += flag[r];
    lfg  += fg ? 1 : 0;
    lbase += base ? 1 : 0;
  }
  sn[t] = lnov; sf[t] = lfg; sb[t] = lbase;
  __syncthreads();
  // Hillis-Steele inclusive scan over novel counts (deterministic ordering)
  for (int off = 1; off < 1024; off <<= 1) {
    int v = (t >= off) ? sn[t - off] : 0;
    __syncthreads();
    sn[t] += v;
    __syncthreads();
  }
  int pos = sn[t] - lnov;  // exclusive prefix
#pragma unroll
  for (int r = 0; r < 8; ++r) {
    if (flag[r]) novel_idx[pos++] = t * 8 + r;
  }
  if (t == 0) {
    int tf = 0, tb = 0;
    for (int i = 0; i < 1024; ++i) { tf += sf[i]; tb += sb[i]; }
    counters[0] = sn[1023];  // n_novel
    counters[1] = tf;        // numel (n_fg)
    counters[2] = tb;        // n_base
  }
}

// ---------------------------------------------------------------------------
// 2) f32 -> (f16 hi, f16 residual) split conversion of features
// ---------------------------------------------------------------------------
__global__ __launch_bounds__(256)
void k_convert(const float* __restrict__ feats, _Float16* __restrict__ Fh,
               _Float16* __restrict__ Fl) {
  const int i = blockIdx.x * blockDim.x + threadIdx.x;
  const float v = feats[i];
  const _Float16 h = (_Float16)v;
  Fh[i] = h;
  Fl[i] = (_Float16)(v - (float)h);
}

// ---------------------------------------------------------------------------
// 3) gather novel rows -> contiguous Fn (zero pad to multiple of 16 rows)
// ---------------------------------------------------------------------------
__global__ __launch_bounds__(128)
void k_gather(const int* __restrict__ counters, const int* __restrict__ novel_idx,
              const _Float16* __restrict__ Fh, const _Float16* __restrict__ Fl,
              _Float16* __restrict__ Gh, _Float16* __restrict__ Gl) {
  const int p = blockIdx.x;
  const int k = threadIdx.x;
  const int nn  = counters[0];
  const int pad = (nn + 15) & ~15;
  if (p < nn) {
    const int src = novel_idx[p];
    Gh[(size_t)p * K_DIM + k] = Fh[(size_t)src * K_DIM + k];
    Gl[(size_t)p * K_DIM + k] = Fl[(size_t)src * K_DIM + k];
  } else if (p < pad) {
    Gh[(size_t)p * K_DIM + k] = (_Float16)0.0f;
    Gl[(size_t)p * K_DIM + k] = (_Float16)0.0f;
  }
}

// ---------------------------------------------------------------------------
// 4) fused gram + row reductions. One wave per 16-row tile (512 waves total).
//    sim tile = (Ah*Bh + Ah*Bl + Al*Bh) * INV_T  (split-f16 ~f32 accuracy)
//    E[i] = sum over novel cols j of exp(sim_ij)   (incl. diag, incl. pads)
//    S[i] = sum over novel cols j of sim_ij
// ---------------------------------------------------------------------------
__global__ __launch_bounds__(256)
void k_gram(const _Float16* __restrict__ Fh, const _Float16* __restrict__ Fl,
            const _Float16* __restrict__ Gh, const _Float16* __restrict__ Gl,
            const int* __restrict__ counters,
            float* __restrict__ E, float* __restrict__ S) {
  const int wave = (blockIdx.x * blockDim.x + threadIdx.x) >> 5;
  const int lane = threadIdx.x & 31;
  const int row0 = wave * 16;
  const int m    = lane & 15;   // A/B row within tile, also D column
  const int hi   = lane >> 4;   // lane-half selector

  const int nn = counters[0];
  const int ntiles = (nn + 15) >> 4;

  // Resident A fragments: 4 chunks of K=32, hi + residual halves.
  // ISA 16-bit A layout: lane (m, half) holds K = {hi*8 + 0..7, 16 + hi*8 + 0..7}
  // per chunk -> two contiguous 16B loads.
  v16h Ah[4], Al[4];
  const _Float16* arow_h = Fh + (size_t)(row0 + m) * K_DIM;
  const _Float16* arow_l = Fl + (size_t)(row0 + m) * K_DIM;
#pragma unroll
  for (int c = 0; c < 4; ++c) {
    const int k0 = c * 32;
    v8h ah0 = *(const v8h*)(arow_h + k0 + hi * 8);
    v8h ah1 = *(const v8h*)(arow_h + k0 + 16 + hi * 8);
    v8h al0 = *(const v8h*)(arow_l + k0 + hi * 8);
    v8h al1 = *(const v8h*)(arow_l + k0 + 16 + hi * 8);
#pragma unroll
    for (int j = 0; j < 8; ++j) {
      Ah[c][j] = ah0[j]; Ah[c][j + 8] = ah1[j];
      Al[c][j] = al0[j]; Al[c][j + 8] = al1[j];
    }
  }

  v8f accE = {};
  v8f accS = {};

  for (int ct = 0; ct < ntiles; ++ct) {
    v8f acc = {};
    // B layout (matches A): lane (n=m, half) holds K = {hi*8 + 0..7, 16 + hi*8 + 0..7}
    // per chunk -> two 16B loads.
    const _Float16* brow_h = Gh + (size_t)(ct * 16 + m) * K_DIM + hi * 8;
    const _Float16* brow_l = Gl + (size_t)(ct * 16 + m) * K_DIM + hi * 8;
#pragma unroll
    for (int c = 0; c < 4; ++c) {
      const int k0 = c * 32;
      v8h bh0 = *(const v8h*)(brow_h + k0);
      v8h bh1 = *(const v8h*)(brow_h + k0 + 16);
      v8h bl0 = *(const v8h*)(brow_l + k0);
      v8h bl1 = *(const v8h*)(brow_l + k0 + 16);
      v16h Bh, Bl;
#pragma unroll
      for (int j = 0; j < 8; ++j) {
        Bh[j] = bh0[j]; Bh[j + 8] = bh1[j];
        Bl[j] = bl0[j]; Bl[j + 8] = bl1[j];
      }
      acc = __builtin_amdgcn_wmma_f32_16x16x32_f16(false, Ah[c], false, Bh,
                                                   (short)0, acc, false, false);
      acc = __builtin_amdgcn_wmma_f32_16x16x32_f16(false, Ah[c], false, Bl,
                                                   (short)0, acc, false, false);
      acc = __builtin_amdgcn_wmma_f32_16x16x32_f16(false, Al[c], false, Bh,
                                                   (short)0, acc, false, false);
    }
    // D layout: component v <-> row (v + hi*8), lane&15 <-> column.
#pragma unroll
    for (int v = 0; v < 8; ++v) {
      const float s = acc[v] * INV_T;
      accE[v] += __expf(s);
      accS[v] += s;
    }
  }

  // Reduce across the 16 lanes of each half (xor butterfly, deterministic).
#pragma unroll
  for (int v = 0; v < 8; ++v) {
    float e  = accE[v];
    float s2 = accS[v];
#pragma unroll
    for (int d = 8; d >= 1; d >>= 1) {
      e  += __shfl_xor(e, d, 32);
      s2 += __shfl_xor(s2, d, 32);
    }
    if (lane == 0)  { E[row0 + v] = e;      S[row0 + v] = s2; }
    if (lane == 16) { E[row0 + 8 + v] = e;  S[row0 + 8 + v] = s2; }
  }
}

// ---------------------------------------------------------------------------
// 5) per-row epilogue: prototype terms, diagonal/pad correction, loss terms
// ---------------------------------------------------------------------------
__global__ __launch_bounds__(256)
void k_rowcombine(const float* __restrict__ feats, const int* __restrict__ labels,
                  const float* __restrict__ protos, const int* __restrict__ counters,
                  const char* __restrict__ nf, const char* __restrict__ bf,
                  const float* __restrict__ E, const float* __restrict__ S,
                  float* __restrict__ cN, float* __restrict__ cB) {
  const int i = blockIdx.x * blockDim.x + threadIdx.x;
  if (i >= M_ROWS) return;
  const float* f = feats + (size_t)i * K_DIM;
  float fr[K_DIM];
#pragma unroll
  for (int k = 0; k < K_DIM; ++k) fr[k] = f[k];

  float sii = 0.f;
#pragma unroll
  for (int k = 0; k < K_DIM; ++k) sii += fr[k] * fr[k];
  sii *= INV_T;

  const int lab = labels[i];
  const int sl = lab < 0 ? 0 : (lab > B_PROTOS - 1 ? B_PROTOS - 1 : lab);
  float spk = 0.f, sepk = 0.f, pi = 0.f;
  for (int b = 0; b < B_PROTOS; ++b) {
    const float* p = protos + (size_t)b * K_DIM;
    float d = 0.f;
#pragma unroll 8
    for (int k = 0; k < K_DIM; ++k) d += fr[k] * p[k];
    d *= INV_T;
    spk  += d;
    sepk += __expf(d);
    if (b == sl) pi = d;
  }

  const int nn  = counters[0];
  const int pad = (nn + 15) & ~15;
  const float Ei = E[i] - (float)(pad - nn);  // pad rows contributed exp(0)=1 each
  const float Si = S[i];                      // pad rows contributed 0

  float outN = 0.f, outB = 0.f;
  if (nf[i]) {
    const float Eoff = Ei - __expf(sii);  // remove diagonal term
    const float Soff = Si - sii;
    const float dn   = Eoff + spk;        // raw pk sum (faithful to source)
    const float nnf  = (float)nn;
    const float cntv = fmaxf(nnf - 1.f, 1.f);
    outN = (-Soff + __logf(dn) * (nnf - 1.f)) / cntv;
  }
  if (bf[i]) {
    const float db = Ei + sepk;           // diagonal excluded automatically (i not novel)
    outB = -(pi - __logf(db));
  }
  cN[i] = outN;
  cB[i] = outB;
}

// ---------------------------------------------------------------------------
// 6) fixed-order final reduction + gating
// ---------------------------------------------------------------------------
__global__ __launch_bounds__(256)
void k_final(const float* __restrict__ cN, const float* __restrict__ cB,
             const int* __restrict__ counters, float* __restrict__ out) {
  __shared__ float rN[256];
  __shared__ float rB[256];
  const int t = threadIdx.x;
  float aN = 0.f, aB = 0.f;
  for (int i = t; i < M_ROWS; i += 256) { aN += cN[i]; aB += cB[i]; }
  rN[t] = aN; rB[t] = aB;
  __syncthreads();
  for (int off = 128; off > 0; off >>= 1) {
    if (t < off) { rN[t] += rN[t + off]; rB[t] += rB[t + off]; }
    __syncthreads();
  }
  if (t == 0) {
    const int nn = counters[0];
    const int numel = counters[1];
    const int nb = counters[2];
    const float ln = (nn > 1) ? rN[0] : 0.f;
    const float lb = (nb > 0) ? rB[0] : 0.f;
    out[0] = (ln + lb) / (float)numel;
  }
}

// ---------------------------------------------------------------------------
extern "C" void kernel_launch(void* const* d_in, const int* in_sizes, int n_in,
                              void* d_out, int out_size, void* d_ws, size_t ws_size,
                              hipStream_t stream) {
  (void)in_sizes; (void)n_in; (void)out_size; (void)ws_size;
  const float* feats  = (const float*)d_in[0];
  const int*   labels = (const int*)d_in[1];
  const float* protos = (const float*)d_in[2];
  // d_in[3] (proto_labels) is arange(B) by construction; encoded as lab < B.
  float* out = (float*)d_out;

  char* ws = (char*)d_ws;
  int*       counters = (int*)(ws + OFF_CNT);
  int*       idx      = (int*)(ws + OFF_IDX);
  char*      nf       = ws + OFF_NF;
  char*      bf       = ws + OFF_BF;
  float*     E        = (float*)(ws + OFF_E);
  float*     S        = (float*)(ws + OFF_S);
  float*     cN       = (float*)(ws + OFF_CN);
  float*     cB       = (float*)(ws + OFF_CB);
  _Float16*  Fh       = (_Float16*)(ws + OFF_FH);
  _Float16*  Fl       = (_Float16*)(ws + OFF_FL);
  _Float16*  Gh       = (_Float16*)(ws + OFF_GH);
  _Float16*  Gl       = (_Float16*)(ws + OFF_GL);

  k_flags<<<1, 1024, 0, stream>>>(labels, counters, idx, nf, bf);
  k_convert<<<(M_ROWS * K_DIM) / 256, 256, 0, stream>>>(feats, Fh, Fl);
  k_gather<<<M_ROWS, K_DIM, 0, stream>>>(counters, idx, Fh, Fl, Gh, Gl);
  k_gram<<<64, 256, 0, stream>>>(Fh, Fl, Gh, Gl, counters, E, S);
  k_rowcombine<<<M_ROWS / 256, 256, 0, stream>>>(feats, labels, protos, counters,
                                                 nf, bf, E, S, cN, cB);
  k_final<<<1, 256, 0, stream>>>(cN, cB, counters, out);
}
